// ConvFC_86053964743071
// MI455X (gfx1250) — matrix operand order repaired and checked by reading, not compiled, not run on hardware
//
#include <hip/hip_runtime.h>
#include <hip/hip_bf16.h>
#include <stdint.h>

// ---------------- problem constants (from the reference) ----------------
#define ENS   10
#define U     256     // N_UNITS
#define NHID  7
#define IND   6
#define OUTD  201
#define BATCH 16384

// ---------------- tiling ----------------
#define MTILE 128              // batch rows per workgroup
#define NTHREADS 512           // 16 waves
#define WSTR  264              // LDS row stride (bf16 elems): 528B = 132 dwords -> 4-bank skew, 16B aligned
#define HSTR  264

typedef __attribute__((ext_vector_type(16))) __bf16 v16bf;
typedef __attribute__((ext_vector_type(8)))  float  v8f;

union FragBF {
  v16bf bf;
  uint4 u[2];
};

static __device__ __forceinline__ unsigned short f2bf(float f) {
  uint32_t u = __float_as_uint(f);
  uint32_t r = (u + 0x7FFFu + ((u >> 16) & 1u)) >> 16;   // round-to-nearest-even
  return (unsigned short)r;
}

__global__ void zero_out_kernel(float* __restrict__ p, int n) {
  int i = blockIdx.x * blockDim.x + threadIdx.x;
  if (i < n) p[i] = 0.0f;
}

// One workgroup: 128 batch rows x one ensemble member, all 9 layers fused.
// Activations live in LDS (bf16); weights staged per-layer (transposed bf16).
__global__ __launch_bounds__(NTHREADS)
void ens_mlp_kernel(const float* __restrict__ x,
                    const float* __restrict__ W0,
                    const float* __restrict__ b0,
                    const float* __restrict__ Wh,
                    const float* __restrict__ bh,
                    const float* __restrict__ Wf,
                    const float* __restrict__ bfin,
                    float* __restrict__ out)
{
  extern __shared__ unsigned short smem[];
  unsigned short* ldsW = smem;                          // U * WSTR bf16  (transposed weights, n-major)
  unsigned short* ldsH = smem + U * WSTR;               // MTILE * HSTR bf16 (activations)
  float*          ldsB = (float*)(smem + U * WSTR + MTILE * HSTR);   // U bias floats

  const int t     = threadIdx.x;
  const int e     = blockIdx.y;
  const int mbase = blockIdx.x * MTILE;

  const int lane  = t & 31;
  const int wave  = t >> 5;
  const int lo    = lane & 15;
  const int hi    = lane >> 4;
  const int msub  = wave >> 1;   // 0..7 : 16-row subtile
  const int nhalf = wave & 1;    // 0..1 : 128-col half

  // ---------------- Layer 0: (128x6) @ W0[e](6x256) + b0, ReLU -> ldsH ----------------
  {
    const int m   = t >> 2;       // 0..127
    const int q   = t & 3;        // column quadrant
    const int row = mbase + m;
    float xv[IND];
#pragma unroll
    for (int i = 0; i < IND; ++i) xv[i] = x[row * IND + i];
    const float* w0e = W0 + (size_t)e * IND * U;
    const float* b0e = b0 + (size_t)e * U;
#pragma unroll 4
    for (int uu = 0; uu < 64; ++uu) {
      const int n = q * 64 + uu;
      float s = b0e[n];
#pragma unroll
      for (int i = 0; i < IND; ++i) s += xv[i] * w0e[i * U + n];
      s = s > 0.0f ? s : 0.0f;
      ldsH[m * HSTR + n] = f2bf(s);
    }
  }
  __syncthreads();

  // ---------------- Layers 1..7 (hidden, ReLU) and 8 (final, atomic-mix out) ----------------
  for (int layer = 0; layer <= NHID; ++layer) {
    const bool is_final = (layer == NHID);

    // ---- stage weights transposed (ldsW[n][k] = bf16(W[k][n])) + bias ----
    // Scalar b32 loads: lanes take consecutive n -> coalesced global reads AND
    // transpose stores land on 16 distinct banks (stride 132 dwords, ~2-way).
    if (!is_final) {
      const float* Wg = Wh + ((size_t)layer * ENS + e) * (U * U);
      const float* bg = bh + ((size_t)layer * ENS + e) * U;
      for (int i = t; i < U * U; i += NTHREADS) {
        const int k = i >> 8;      // 0..255 (row of W = input unit)
        const int n = i & 255;     // 0..255 (col of W = output unit), lane-consecutive
        ldsW[n * WSTR + k] = f2bf(Wg[i]);
      }
      if (t < U) ldsB[t] = bg[t];
    } else {
      const float* Wg = Wf + (size_t)e * (U * OUTD);
      const float* bg = bfin + (size_t)e * OUTD;
      for (int i = t; i < U * U; i += NTHREADS) {
        const int k = i >> 8;
        const int n = i & 255;
        const float v = (n < OUTD) ? Wg[k * OUTD + n] : 0.0f;   // zero-pad 201 -> 256
        ldsW[n * WSTR + k] = f2bf(v);
      }
      if (t < U) ldsB[t] = (t < OUTD) ? bg[t] : 0.0f;
    }

    // warm L2 with next layer's weights while this layer computes
    {
      const float* nxt = (layer + 1 < NHID)
                             ? (Wh + ((size_t)(layer + 1) * ENS + e) * (U * U))
                             : ((layer + 1 == NHID) ? (Wf + (size_t)e * (U * OUTD)) : nullptr);
      if (nxt) __builtin_prefetch(nxt + t * 128, 0, 1);
    }
    __syncthreads();

    // ---- GEMM: rows [msub*16, +16), cols [nhalf*128, +128) = 8 accum tiles ----
    const unsigned short* hrow = &ldsH[(msub * 16 + lo) * HSTR + hi * 16];
    const unsigned short* wrow = &ldsW[(nhalf * 128 + lo) * WSTR + hi * 16];

    v8f zero = {};
    v8f acc[8];
#pragma unroll
    for (int i = 0; i < 8; ++i) acc[i] = zero;

#pragma unroll
    for (int kk = 0; kk < 8; ++kk) {               // K = 256 in steps of 32
      // A fragment 16x32 bf16: row lo, K = kk*32 + (hi*16..+15) (contiguous, matches B)
      FragBF a;
      {
        const unsigned short* ah = hrow + kk * 32;
        a.u[0] = *(const uint4*)(ah);
        a.u[1] = *(const uint4*)(ah + 8);          // +16B: K offset +8
      }
      // B fragments ping-pong double-buffered so each WMMA's operands are
      // loaded one WMMA ahead (partial s_wait_dscnt instead of full drain).
      FragBF b[2];
      {
        const unsigned short* bw = wrow + kk * 32; // nt = 0
        b[0].u[0] = *(const uint4*)(bw);
        b[0].u[1] = *(const uint4*)(bw + 8);       // +16B
      }
#pragma unroll
      for (int nt = 0; nt < 8; ++nt) {
        if (nt + 1 < 8) {
          const unsigned short* bw = wrow + (nt + 1) * 16 * WSTR + kk * 32;
          b[(nt + 1) & 1].u[0] = *(const uint4*)(bw);
          b[(nt + 1) & 1].u[1] = *(const uint4*)(bw + 8);
        }
        acc[nt] = __builtin_amdgcn_wmma_f32_16x16x32_bf16(
            false, a.bf, false, b[nt & 1].bf, (short)0, acc[nt], false, false);
      }
    }
    __syncthreads();   // all ldsH/ldsW reads done before anyone overwrites them

    if (!is_final) {
      // bias + ReLU, write next activations (C/D layout: VGPR r -> M = r + hi*8, N = lo)
#pragma unroll
      for (int nt = 0; nt < 8; ++nt) {
        const int n    = nhalf * 128 + nt * 16 + lo;
        const float bb = ldsB[n];
#pragma unroll
        for (int r = 0; r < 8; ++r) {
          float v = acc[nt][r] + bb;
          v = v > 0.0f ? v : 0.0f;
          const int m = msub * 16 + hi * 8 + r;
          ldsH[m * HSTR + n] = f2bf(v);
        }
      }
      __syncthreads();
    } else {
      // final layer: y[b,e,n] contributes once to out[b, (e*201+n)/10] * (1/10)
#pragma unroll
      for (int nt = 0; nt < 8; ++nt) {
        const int n = nhalf * 128 + nt * 16 + lo;
        if (n < OUTD) {
          const float bb   = ldsB[n];
          const int   flat = e * OUTD + n;
          const int   o    = flat / ENS;
#pragma unroll
          for (int r = 0; r < 8; ++r) {
            const float v = acc[nt][r] + bb;
            const int   m = msub * 16 + hi * 8 + r;
            atomicAdd(&out[(size_t)(mbase + m) * OUTD + o], v * (1.0f / ENS));
          }
        }
      }
    }
  }
}

extern "C" void kernel_launch(void* const* d_in, const int* in_sizes, int n_in,
                              void* d_out, int out_size, void* d_ws, size_t ws_size,
                              hipStream_t stream) {
  const float* x   = (const float*)d_in[0];
  const float* W0  = (const float*)d_in[1];
  const float* b0  = (const float*)d_in[2];
  const float* Wh  = (const float*)d_in[3];
  const float* bh  = (const float*)d_in[4];
  const float* Wf  = (const float*)d_in[5];
  const float* bfp = (const float*)d_in[6];
  float* out = (float*)d_out;

  const int n = BATCH * OUTD;
  zero_out_kernel<<<(n + 255) / 256, 256, 0, stream>>>(out, n);

  const size_t lds_bytes =
      (size_t)(U * WSTR + MTILE * HSTR) * sizeof(unsigned short) + U * sizeof(float);
  dim3 grid(BATCH / MTILE, ENS);
  ens_mlp_kernel<<<grid, dim3(NTHREADS), lds_bytes, stream>>>(x, W0, b0, Wh, bh, Wf, bfp, out);
}
